// GNN_Prediction_Model_66752381714446
// MI455X (gfx1250) — matrix operand order repaired and checked by reading, not compiled, not run on hardware
//
#include <hip/hip_runtime.h>
#include <hip/hip_bf16.h>

// MI455X (gfx1250) implementation.
// 176 GFLOP total, all operands L2-resident (Q/K/V 8MB each, W1T 4MB, L2=192MB)
// -> matrix-pipe bound. All four GEMMs run on v_wmma_f32_16x16x32_f16 (f32 acc).
// Flash-attention streaming without max-rescale (scores bounded ~|8| -> exp is
// f32-safe), so the 16x512 output tile stays in WMMA C registers (64 VGPRs).
// LDS tile staging is done by the Tensor Data Mover (tensor_load_to_lds with
// D# pad fields producing the bank-staggered LDS pitches), synced on TENSORcnt.

typedef __attribute__((ext_vector_type(16))) _Float16 v16h;
typedef __attribute__((ext_vector_type(8)))  _Float16 v8h;
typedef __attribute__((ext_vector_type(8)))  float    v8f;
typedef __attribute__((ext_vector_type(4)))  unsigned v4u;
typedef __attribute__((ext_vector_type(8)))  int      v8i;
typedef __attribute__((ext_vector_type(4)))  int      v4i;

#define NTOK 8192
#define CIN  128
#define DD   512
#define HH   4096

// LDS row strides (in elements), chosen 16B-aligned + bank-staggered (64 banks)
constexpr int KSTR = 520;   // halves: rows of 512 halves + 16B pad (260 dwords)
constexpr int VSTR = 72;    // halves: rows of 64 halves + 16B pad (36 dwords)
constexpr int PSTR = 72;    // halves
constexpr int OSTR = 152;   // halves: rows of 128 halves + 48B pad (76 dwords)
constexpr int HSTR = 4100;  // floats

constexpr size_t QKV_SMEM  = size_t(16) * OSTR * 2;
constexpr size_t ATTN_SMEM = size_t(32 + 64) * KSTR * 2 + size_t(512) * VSTR * 2
                           + size_t(2) * 16 * PSTR * 2 + 2 * 16 * 4;
constexpr size_t MLP_SMEM  = size_t(16) * KSTR * 2 + size_t(16) * HSTR * 4 + 2 * 16 * 4;

__device__ __forceinline__ v8f wmma_f16(v16h a, v16h b, v8f c) {
  return __builtin_amdgcn_wmma_f32_16x16x32_f16(false, a, false, b, (short)0, c,
                                                false, false);
}

// A-operand (16xK row-striped, 16-bit): lane holds M=lane&15; half=(lane>>4):
// halves [kk0+8*half .. +7] and [kk0+16+8*half .. +7]  (two 16B chunks).
__device__ __forceinline__ v16h ldA(const _Float16* p) {
  v8h lo = *(const v8h*)p;
  v8h hi = *(const v8h*)(p + 16);
  v16h r;
#pragma unroll
  for (int i = 0; i < 8; ++i) { r[i] = lo[i]; r[i + 8] = hi[i]; }
  return r;
}
// B-operand (Kx16, 16-bit): lane n=lane&15; half=(lane>>4):
// halves [kk0+8*half .. +7] and [kk0+16+8*half .. +7]  (two 16B chunks).
__device__ __forceinline__ v16h ldB(const _Float16* p) {
  v8h lo = *(const v8h*)p;
  v8h hi = *(const v8h*)(p + 16);
  v16h r;
#pragma unroll
  for (int i = 0; i < 8; ++i) { r[i] = lo[i]; r[i + 8] = hi[i]; }
  return r;
}

// ---- Tensor Data Mover: 2D tile (16-bit elements) global -> LDS with row pad.
// D# per CDNA5 ISA ch8: group0 = {flags(count=1), lds_addr, gaddr_lo,
// gaddr_hi|type=2}; group1 = data_size=1(2B), pad_enable, pad_interval code pi
// (2^(pi+1) dwords), pad_amount code pa (pa+1 dwords), dims/strides/tile dims.
// Flat LDS pointers: aperture doc says addr[31:0] IS the LDS byte address.
__device__ __forceinline__ void tdm_load_2d(const void* g, void* l,
                                            unsigned d0, unsigned d1,
                                            unsigned long long stride_elems,
                                            unsigned tile_d0, unsigned tile_d1,
                                            unsigned pi, unsigned pa) {
  unsigned long long ga = (unsigned long long)(uintptr_t)g;
  v4u g0 = {1u, (unsigned)(uintptr_t)l, (unsigned)ga,
            (unsigned)((ga >> 32) & 0x1FFFFFFull) | (2u << 30)};
  v8i g1;
  g1[0] = (int)((1u << 16) | (1u << 20) | (pi << 22) | (pa << 25));
  g1[1] = (int)((d0 & 0xFFFFu) << 16);
  g1[2] = (int)(((d0 >> 16) & 0xFFFFu) | ((d1 & 0xFFFFu) << 16));
  g1[3] = (int)(((d1 >> 16) & 0xFFFFu) | ((tile_d0 & 0xFFFFu) << 16));
  g1[4] = (int)(tile_d1 & 0xFFFFu);
  g1[5] = (int)(unsigned)(stride_elems & 0xFFFFFFFFull);
  g1[6] = (int)(unsigned)((stride_elems >> 32) & 0xFFFFull);
  g1[7] = 0;
  v4i z4 = {};
  v8i z8 = {};
  // clang-23 / therock signature: (v4u, v8i, v4i, v4i, v8i, i32 cpol)
  __builtin_amdgcn_tensor_load_to_lds(g0, g1, z4, z4, z8, 0);
}

// ---------------- prep kernels ----------------
__global__ void cvt_f16_kernel(const float* __restrict__ in,
                               _Float16* __restrict__ out, int n) {
  int i = blockIdx.x * 256 + threadIdx.x;
  if (i < n) out[i] = (_Float16)in[i];
}
// out[c*rows + r] = in[r*cols + c]   (rows x cols -> cols x rows, f32 -> f16)
__global__ void transpose_f16_kernel(const float* __restrict__ in,
                                     _Float16* __restrict__ out, int rows, int cols) {
  int i = blockIdx.x * 256 + threadIdx.x;
  if (i < rows * cols) {
    int r = i / cols, c = i - r * cols;
    out[c * rows + r] = (_Float16)in[i];
  }
}

// ---------------- QKV projection ----------------
// grid = (512 row-tiles, 3 matrices); blockIdx.y selects q/k/v so the epilogue
// branch is scalar (SGPR) flow, not per-lane cndmask. 8 waves x 4 n-tiles.
__global__ __launch_bounds__(256) void qkv_kernel(
    const _Float16* __restrict__ obs16, const _Float16* __restrict__ WqT,
    const _Float16* __restrict__ WkT, const _Float16* __restrict__ WvT,
    const float* __restrict__ bq, const float* __restrict__ bk,
    const float* __restrict__ bv, _Float16* __restrict__ q16,
    _Float16* __restrict__ k16, _Float16* __restrict__ vT16) {
  extern __shared__ char smem[];
  _Float16* Ot = (_Float16*)smem;  // 16 x OSTR
  const int t = threadIdx.x, lane = t & 31, wid = t >> 5;
  const int half = (lane >> 4) & 1, l15 = lane & 15;
  const int rowbase = blockIdx.x * 16;
  const int mat = blockIdx.y;  // scalar: 0=q, 1=k, 2=v

  if (wid == 0) {
    // 16x128 tile, contiguous rows; pad 48B after every 128 halves (256B=64dw)
    tdm_load_2d(obs16 + (size_t)rowbase * CIN, Ot, CIN, 16, CIN, CIN, 16,
                /*pi=*/5, /*pa=*/11);
    __builtin_amdgcn_s_wait_tensorcnt(0);
  }
  __syncthreads();

  const _Float16* WT = (mat == 0) ? WqT : (mat == 1) ? WkT : WvT;
  const float* bias = (mat == 0) ? bq : (mat == 1) ? bk : bv;

  const _Float16* arow = Ot + l15 * OSTR + 8 * half;
  v16h A[4];
#pragma unroll
  for (int s = 0; s < 4; ++s) A[s] = ldA(arow + 32 * s);

  const float qscale = 0.04419417382415922f;  // 1/sqrt(512)
#pragma unroll
  for (int i = 0; i < 4; ++i) {
    int nt = wid * 4 + i;  // 0..31
    const _Float16* brow = WT + (size_t)(16 * nt + l15) * CIN + 8 * half;
    v8f acc = {};
#pragma unroll
    for (int s = 0; s < 4; ++s) acc = wmma_f16(A[s], ldB(brow + 32 * s), acc);
    int col = 16 * nt + l15;
    float bb = bias[col];
    if (mat == 0) {
#pragma unroll
      for (int j = 0; j < 8; ++j) {
        int row = rowbase + j + 8 * half;
        q16[(size_t)row * DD + col] = (_Float16)((acc[j] + bb) * qscale);
      }
    } else if (mat == 1) {
#pragma unroll
      for (int j = 0; j < 8; ++j) {
        int row = rowbase + j + 8 * half;
        k16[(size_t)row * DD + col] = (_Float16)(acc[j] + bb);
      }
    } else {
#pragma unroll
      for (int j = 0; j < 8; ++j) {
        int row = rowbase + j + 8 * half;
        vT16[(size_t)col * NTOK + row] = (_Float16)(acc[j] + bb);  // transposed V
      }
    }
  }
}

// ---------------- flash attention (no max-rescale) ----------------
// grid.x = 256 WGs (32 q-rows each); 8 waves: wave w -> row-tile r=w>>2,
// quarter qd=w&3. 64-token K/V blocks staged to LDS by the TDM.
__global__ __launch_bounds__(256) void attn_kernel(
    const _Float16* __restrict__ q16, const _Float16* __restrict__ k16,
    const _Float16* __restrict__ vT16, _Float16* __restrict__ out16) {
  extern __shared__ char smem[];
  _Float16* qT = (_Float16*)smem;            // 32 x KSTR
  _Float16* Kt = qT + 32 * KSTR;             // 64 x KSTR
  _Float16* Vt = Kt + 64 * KSTR;             // 512 x VSTR (d-major, transposed V)
  _Float16* Pt = Vt + 512 * VSTR;            // 2 x 16 x PSTR
  float* lsum = (float*)(Pt + 2 * 16 * PSTR);  // 2 x 16

  const int t = threadIdx.x, lane = t & 31, wid = t >> 5;
  const int half = (lane >> 4) & 1, l15 = lane & 15;
  const int r = wid >> 2, qd = wid & 3;
  const int rowbase = blockIdx.x * 32;

  if (wid == 0) {
    // q tile 32x512, contiguous; pad 16B per 512-half row -> KSTR pitch
    tdm_load_2d(q16 + (size_t)rowbase * DD, qT, DD, 32, DD, DD, 32,
                /*pi=*/7, /*pa=*/3);
  }
  if (t < 32) lsum[t] = 0.f;

  v8f acc[8];
  v8f vzero = {};
#pragma unroll
  for (int i = 0; i < 8; ++i) acc[i] = vzero;
  float lpart[8] = {0.f, 0.f, 0.f, 0.f, 0.f, 0.f, 0.f, 0.f};

  const _Float16* arow = qT + (16 * r + l15) * KSTR + 8 * half;
  const _Float16* prow = Pt + (r * 16 + l15) * PSTR + 8 * half;
  const _Float16* vbase = Vt + (128 * qd + l15) * VSTR + 8 * half;

  for (int jb = 0; jb < NTOK / 64; ++jb) {
    const int j0 = jb * 64;
    __syncthreads();  // previous block's S/P@V reads done
    if (wid == 0) {
      // K: 64x512 contiguous rows -> KSTR pitch (pad 4dw per 256dw)
      tdm_load_2d(k16 + (size_t)j0 * DD, Kt, DD, 64, DD, DD, 64,
                  /*pi=*/7, /*pa=*/3);
      // V^T: 512 rows x 64 halves, row stride NTOK -> VSTR pitch (pad 4dw/32dw)
      tdm_load_2d(vT16 + j0, Vt, NTOK, DD, NTOK, 64, DD,
                  /*pi=*/4, /*pa=*/3);
      __builtin_amdgcn_s_wait_tensorcnt(0);
    }
    __syncthreads();

    // S(16x16) = q_tile(r) @ K^T for tokens [j0+16*qd, +16)
    v8f S = vzero;
    const _Float16* brow = Kt + (16 * qd + l15) * KSTR + 8 * half;
#pragma unroll
    for (int s = 0; s < 16; ++s)
      S = wmma_f16(ldA(arow + 32 * s), ldB(brow + 32 * s), S);

    // exp (1/sqrt(D) folded into q16), row-sums, pack P to LDS
#pragma unroll
    for (int i = 0; i < 8; ++i) {
      float e = __expf(S[i]);
      float sum = e;
#pragma unroll
      for (int m = 1; m < 16; m <<= 1) sum += __shfl_xor(sum, m, 16);
      lpart[i] += sum;
      Pt[(r * 16 + i + 8 * half) * PSTR + 16 * qd + l15] = (_Float16)e;
    }
    __syncthreads();

    // acc(r, 128-col chunk qd) += P(16x64) @ V(64x128)
    v16h A0 = ldA(prow);
    v16h A1 = ldA(prow + 32);
#pragma unroll
    for (int nt = 0; nt < 8; ++nt) {
      const _Float16* vb = vbase + nt * 16 * VSTR;
      acc[nt] = wmma_f16(A0, ldB(vb), acc[nt]);
      acc[nt] = wmma_f16(A1, ldB(vb + 32), acc[nt]);
    }
  }

  // combine row-sums across the 4 quarter-waves of each row-tile
  if (l15 == 0) {
#pragma unroll
    for (int i = 0; i < 8; ++i)
      atomicAdd(&lsum[r * 16 + i + 8 * half], lpart[i]);
  }
  __syncthreads();

  float rinv[8];
#pragma unroll
  for (int i = 0; i < 8; ++i) rinv[i] = 1.f / lsum[r * 16 + i + 8 * half];
#pragma unroll
  for (int nt = 0; nt < 8; ++nt) {
    int col = 128 * qd + 16 * nt + l15;
#pragma unroll
    for (int i = 0; i < 8; ++i) {
      int row = rowbase + 16 * r + i + 8 * half;
      out16[(size_t)row * DD + col] = (_Float16)(acc[nt][i] * rinv[i]);
    }
  }
}

// ---------------- MLP + layernorm + head + ODE ----------------
// grid.x = 512 WGs of 16 rows. 8 waves each build a 512-wide H chunk into a
// 262KB LDS h-tile (single pass thanks to 320KB WGP LDS), then layernorm+relu+
// W2 head + 100-step bicycle model in f32 VALU.
__global__ __launch_bounds__(256) void mlp_kernel(
    const _Float16* __restrict__ out16, const _Float16* __restrict__ W1T,
    const float* __restrict__ b1, const float* __restrict__ g1,
    const float* __restrict__ beta1, const float* __restrict__ W2,
    const float* __restrict__ b2, const float* __restrict__ obs,
    float* __restrict__ pred) {
  extern __shared__ char smem[];
  _Float16* Xt = (_Float16*)smem;          // 16 x KSTR
  float* Hs = (float*)(Xt + 16 * KSTR);    // 16 x HSTR
  float* thr_s = Hs + 16 * HSTR;           // 16
  float* del_s = thr_s + 16;               // 16

  const int t = threadIdx.x, lane = t & 31, wid = t >> 5;
  const int half = (lane >> 4) & 1, l15 = lane & 15;
  const int rowbase = blockIdx.x * 16;

  if (wid == 0) {
    tdm_load_2d(out16 + (size_t)rowbase * DD, Xt, DD, 16, DD, DD, 16,
                /*pi=*/7, /*pa=*/3);
    __builtin_amdgcn_s_wait_tensorcnt(0);
  }
  __syncthreads();

  const _Float16* arow = Xt + l15 * KSTR + 8 * half;
  for (int nt = 0; nt < 32; ++nt) {
    int col = 512 * wid + 16 * nt + l15;
    const _Float16* brow = W1T + (size_t)col * DD + 8 * half;
    v8f acc = {};
#pragma unroll
    for (int s = 0; s < 16; ++s)
      acc = wmma_f16(ldA(arow + 32 * s), ldB(brow + 32 * s), acc);
    float bb = b1[col];
#pragma unroll
    for (int j = 0; j < 8; ++j) Hs[(j + 8 * half) * HSTR + col] = acc[j] + bb;
  }
  __syncthreads();

  // layernorm + relu + W2 head: 16 groups of 16 threads, one row each
  int g = t >> 4, i = t & 15;
  float sum = 0.f, sq = 0.f;
  for (int c = i; c < HH; c += 16) {
    float h = Hs[g * HSTR + c];
    sum += h; sq += h * h;
  }
#pragma unroll
  for (int m = 1; m < 16; m <<= 1) {
    sum += __shfl_xor(sum, m, 16);
    sq += __shfl_xor(sq, m, 16);
  }
  float mu = sum * (1.f / HH);
  float var = sq * (1.f / HH) - mu * mu;
  float rs = rsqrtf(var + 1e-5f);
  float thr = 0.f, del = 0.f;
  for (int c = i; c < HH; c += 16) {
    float h = (Hs[g * HSTR + c] - mu) * rs * g1[c] + beta1[c];
    h = h > 0.f ? h : 0.f;
    thr += h * W2[c * 2 + 0];
    del += h * W2[c * 2 + 1];
  }
#pragma unroll
  for (int m = 1; m < 16; m <<= 1) {
    thr += __shfl_xor(thr, m, 16);
    del += __shfl_xor(del, m, 16);
  }
  if (i == 0) { thr_s[g] = thr + b2[0]; del_s[g] = del + b2[1]; }
  __syncthreads();

  // bicycle-model rollout, one thread per row
  if (t < 16) {
    int row = rowbase + t;
    float throttle = thr_s[t], delta = del_s[t];
    const float* o = obs + (size_t)row * CIN;
    float xx = o[0] * 10.f, yy = o[1] * 10.f, yw = o[4] * 10.f;
    float vx = o[2] * 10.f, vy = o[3] * 10.f;
    float vv = sqrtf(vx * vx + vy * vy);
    const float dt = 0.001f;  // DT / DT_DISCRE
    float tk = tanf(delta) * (1.f / 2.96f);
    for (int s = 0; s < 100; ++s) {
      vv += throttle * dt;  // C_R = C_A = 0
      float sn, cs;
      sincosf(yw, &sn, &cs);
      xx += vv * cs * dt;
      yy += vv * sn * dt;
      yw += vv * tk * dt;
      yw = atan2f(sinf(yw), cosf(yw));
    }
    float sn, cs;
    sincosf(yw, &sn, &cs);
    pred[row * 5 + 0] = xx * 0.1f;
    pred[row * 5 + 1] = yy * 0.1f;
    pred[row * 5 + 2] = vv * cs * 0.1f;
    pred[row * 5 + 3] = vv * sn * 0.1f;
    pred[row * 5 + 4] = yw * 0.1f;
  }
}

extern "C" void kernel_launch(void* const* d_in, const int* in_sizes, int n_in,
                              void* d_out, int out_size, void* d_ws, size_t ws_size,
                              hipStream_t stream) {
  const float* obs = (const float*)d_in[0];
  const float* Wq = (const float*)d_in[1];
  const float* bq = (const float*)d_in[2];
  const float* Wk = (const float*)d_in[3];
  const float* bk = (const float*)d_in[4];
  const float* Wv = (const float*)d_in[5];
  const float* bv = (const float*)d_in[6];
  const float* W1 = (const float*)d_in[7];
  const float* b1 = (const float*)d_in[8];
  const float* g1 = (const float*)d_in[9];
  const float* beta1 = (const float*)d_in[10];
  const float* W2 = (const float*)d_in[11];
  const float* b2 = (const float*)d_in[12];
  float* pred = (float*)d_out;

  // workspace carve (f16 buffers), ~38.5MB total
  char* w = (char*)d_ws;
  auto take = [&](size_t elems) {
    _Float16* p = (_Float16*)w;
    w += ((elems * 2 + 255) & ~size_t(255));
    return p;
  };
  _Float16* obs16 = take((size_t)NTOK * CIN);
  _Float16* WqT = take((size_t)DD * CIN);
  _Float16* WkT = take((size_t)DD * CIN);
  _Float16* WvT = take((size_t)DD * CIN);
  _Float16* W1T = take((size_t)HH * DD);
  _Float16* q16 = take((size_t)NTOK * DD);
  _Float16* k16 = take((size_t)NTOK * DD);
  _Float16* vT16 = take((size_t)DD * NTOK);
  _Float16* out16 = take((size_t)NTOK * DD);

  int n;
  n = NTOK * CIN;
  cvt_f16_kernel<<<(n + 255) / 256, 256, 0, stream>>>(obs, obs16, n);
  n = CIN * DD;
  transpose_f16_kernel<<<(n + 255) / 256, 256, 0, stream>>>(Wq, WqT, CIN, DD);
  transpose_f16_kernel<<<(n + 255) / 256, 256, 0, stream>>>(Wk, WkT, CIN, DD);
  transpose_f16_kernel<<<(n + 255) / 256, 256, 0, stream>>>(Wv, WvT, CIN, DD);
  n = DD * HH;
  transpose_f16_kernel<<<(n + 255) / 256, 256, 0, stream>>>(W1, W1T, DD, HH);

  dim3 qkv_grid(NTOK / 16, 3);
  qkv_kernel<<<qkv_grid, 256, QKV_SMEM, stream>>>(obs16, WqT, WkT, WvT, bq, bk,
                                                  bv, q16, k16, vT16);
  attn_kernel<<<NTOK / 32, 256, ATTN_SMEM, stream>>>(q16, k16, vT16, out16);
  mlp_kernel<<<NTOK / 16, 256, MLP_SMEM, stream>>>(out16, W1T, b1, g1, beta1, W2,
                                                   b2, obs, pred);
}
